// LearnedBindingMemory_59974923321458
// MI455X (gfx1250) — matrix operand order repaired and checked by reading, not compiled, not run on hardware
//
#include <hip/hip_runtime.h>

// ---------------------------------------------------------------------------
// MI455X fused LearnedBindingMemory
//   Roofline: 128MB streamed (keys+values) / 23.3 TB/s ~= 5.5us floor.
//   All math linear past h = (K Wbk^T + b) * (V Wbv^T + b) and summed over n,
//   so the kernel reduces to two [32768,512]x[512,64] GEMMs (f16 WMMA, f32
//   accumulate) + an elementwise product + a 64-wide per-batch reduction.
//   Weights are staged once per workgroup into LDS (bank-conflict padded) so
//   the hot loop's B fragments come from ds_load_b128 instead of L2.
// ---------------------------------------------------------------------------

#define DD   512
#define BDIM 64
#define NSEQ 4096
#define NB   8
#define LDSW (DD + 16)   // padded LDS row stride in halves: 264 dwords == 8 mod 64 banks

typedef __attribute__((ext_vector_type(16))) _Float16 v16h;
typedef __attribute__((ext_vector_type(8)))  _Float16 v8h;
typedef __attribute__((ext_vector_type(8)))  float    v8f;

// ---------------- kernel 0: zero accumulators + convert weights to f16 -----
__global__ __launch_bounds__(256) void prep_kernel(
    const float* __restrict__ Wbk, const float* __restrict__ Wbv,
    _Float16* __restrict__ wk16, _Float16* __restrict__ wv16,
    float* __restrict__ S) {
  int idx = blockIdx.x * 256 + threadIdx.x;
  if (idx < NB * BDIM) S[idx] = 0.0f;
  if (idx < BDIM * DD) {
    wk16[idx] = (_Float16)Wbk[idx];
    wv16[idx] = (_Float16)Wbv[idx];
  }
}

// ---------------- kernel 1: WMMA bind + reduce over n ----------------------
// Grid: 256 blocks x 256 threads. Block = 128 rows (8 waves x 16 rows).
// Per wave: acc_k/acc_v = [16 rows x 64 cols] via 4 N-tiles, K looped 512/32.
__global__ __launch_bounds__(256) void bind_reduce_kernel(
    const float* __restrict__ keys, const float* __restrict__ values,
    const _Float16* __restrict__ wk16, const _Float16* __restrict__ wv16,
    const float* __restrict__ bbk, const float* __restrict__ bbv,
    float* __restrict__ S) {
  __shared__ _Float16 sWk[BDIM * LDSW];   // ~67.6 KB
  __shared__ _Float16 sWv[BDIM * LDSW];   // ~67.6 KB  (135 KB total < 320 KB/WGP)
  __shared__ float    red[BDIM];

  const int lane = threadIdx.x & 31;
  const int wave = threadIdx.x >> 5;
  const int m    = lane & 15;       // row within 16-row tile / N within 16-col tile
  const int kh   = lane >> 4;       // lane-half selects K sub-chunk
  const int rowTile = blockIdx.x * 128 + wave * 16;
  const int b       = blockIdx.x >> 5;            // 32 blocks per batch element

  // ---- stage f16 weights into LDS (row-padded), 8-half chunks -------------
  // 4096 chunks per matrix, 16 per thread: global_load_b128 -> ds_store_b128
  for (int i = threadIdx.x; i < (BDIM * DD) / 8; i += 256) {
    const int r = i >> 6;          // 64 chunks of 8 per 512-wide row
    const int c = (i & 63) * 8;
    *(v8h*)(sWk + r * LDSW + c) = *(const v8h*)(wk16 + r * DD + c);
    *(v8h*)(sWv + r * LDSW + c) = *(const v8h*)(wv16 + r * DD + c);
  }
  if (threadIdx.x < BDIM) red[threadIdx.x] = 0.0f;
  __syncthreads();

  const float* kp = keys   + (long)(rowTile + m) * DD;
  const float* vp = values + (long)(rowTile + m) * DD;

  v8f acck[4] = {};
  v8f accv[4] = {};

  for (int ks = 0; ks < DD / 32; ++ks) {
    // ---- A fragments (16x32 f16): lane-half holds K = kh*16..+16
    const int a0 = ks * 32 + kh * 16;
    v8f klo = *(const v8f*)(kp + a0);
    v8f khi = *(const v8f*)(kp + a0 + 8);
    v8f vlo = *(const v8f*)(vp + a0);
    v8f vhi = *(const v8f*)(vp + a0 + 8);
    if (ks + 1 < DD / 32) {               // stream next K-chunk (global_prefetch_b8)
      __builtin_prefetch(kp + a0 + 32, 0, 0);
      __builtin_prefetch(vp + a0 + 32, 0, 0);
    }
    v16h ak, av;
#pragma unroll
    for (int i = 0; i < 8; ++i) {
      ak[i]     = (_Float16)klo[i];
      ak[i + 8] = (_Float16)khi[i];
      av[i]     = (_Float16)vlo[i];
      av[i + 8] = (_Float16)vhi[i];
    }
    // ---- B fragments (32x16 f16) from LDS: lane-half K = kh*16..+16, col = m
    const int bcol = ks * 32 + kh * 16;
#pragma unroll
    for (int nt = 0; nt < 4; ++nt) {
      const int wrow = nt * 16 + m;
      v16h bk = *(const v16h*)(sWk + wrow * LDSW + bcol);
      v16h bv = *(const v16h*)(sWv + wrow * LDSW + bcol);
      acck[nt] = __builtin_amdgcn_wmma_f32_16x16x32_f16(
          false, ak, false, bk, (short)0, acck[nt], false, false);
      accv[nt] = __builtin_amdgcn_wmma_f32_16x16x32_f16(
          false, av, false, bv, (short)0, accv[nt], false, false);
    }
  }

  // ---- h = (k + bbk) * (v + bbv); reduce rows -> per-column partial sums.
  // C/D layout: component j holds row m_out = j + 8*kh, column = lane&15.
  float colsum[4];
#pragma unroll
  for (int nt = 0; nt < 4; ++nt) {
    const int colIdx = nt * 16 + m;
    const float bk = bbk[colIdx];
    const float bv = bbv[colIdx];
    float s = 0.0f;
#pragma unroll
    for (int j = 0; j < 8; ++j)
      s += (acck[nt][j] + bk) * (accv[nt][j] + bv);
    s += __shfl_xor(s, 16, 32);   // combine the two 8-row halves (wave32)
    colsum[nt] = s;
  }

  __syncthreads();                 // red[] initialized before the GEMM loop
  if (lane < 16) {
#pragma unroll
    for (int nt = 0; nt < 4; ++nt)
      atomicAdd(&red[nt * 16 + m], colsum[nt]);      // ds_add_f32
  }
  __syncthreads();
  if (threadIdx.x < BDIM)
    atomicAdd(&S[b * BDIM + threadIdx.x], red[threadIdx.x]);  // global_atomic_add_f32
}

// ---------------- kernel 2: per-batch epilogue down to LayerNorm -----------
__global__ __launch_bounds__(256) void epilogue_kernel(
    const float* __restrict__ S, const float* __restrict__ query,
    const float* __restrict__ Wbc, const float* __restrict__ bbc,
    const float* __restrict__ Wuq, const float* __restrict__ buq,
    const float* __restrict__ Wue, const float* __restrict__ bue,
    const float* __restrict__ ln_g, const float* __restrict__ ln_b,
    float* __restrict__ normed) {
  const int b = blockIdx.x;
  const int t = threadIdx.x;
  const float nF = (float)NSEQ;

  __shared__ float sS[BDIM];
  __shared__ float t1[DD];
  __shared__ float qb[BDIM];
  __shared__ float m2[BDIM];
  __shared__ float ret[DD];
  __shared__ float red[256];

  if (t < BDIM) sS[t] = S[b * BDIM + t];
  __syncthreads();

  // t1 = Wbc @ S_b + n*bbc   (sum over n of bindings)
  for (int d = t; d < DD; d += 256) {
    float acc = 0.0f;
    const float* wr = Wbc + d * BDIM;
#pragma unroll 8
    for (int k = 0; k < BDIM; ++k) acc += wr[k] * sS[k];
    t1[d] = acc + nF * bbc[d];
  }
  // q_b = Wuq @ query_b + buq
  if (t < BDIM) {
    float acc = 0.0f;
    const float* wr = Wuq + t * DD;
    const float* qr = query + b * DD;
    for (int j = 0; j < DD; ++j) acc += wr[j] * qr[j];
    qb[t] = acc + buq[t];
  }
  __syncthreads();
  // m2 = (Wuq @ t1 + n*buq) * q_b   (= sum over n of match)
  if (t < BDIM) {
    float acc = 0.0f;
    const float* wr = Wuq + t * DD;
    for (int j = 0; j < DD; ++j) acc += wr[j] * t1[j];
    m2[t] = (acc + nF * buq[t]) * qb[t];
  }
  __syncthreads();
  // retrieved = (Wue @ m2 + n*bue) / sqrt(n),  sqrt(4096) = 64
  for (int d = t; d < DD; d += 256) {
    float acc = 0.0f;
    const float* wr = Wue + d * BDIM;
#pragma unroll 8
    for (int k = 0; k < BDIM; ++k) acc += wr[k] * m2[k];
    ret[d] = (acc + nF * bue[d]) * (1.0f / 64.0f);
  }
  __syncthreads();

  // LayerNorm (eps 1e-5)
  red[t] = ret[t] + ret[t + 256];
  __syncthreads();
  for (int s2 = 128; s2 > 0; s2 >>= 1) {
    if (t < s2) red[t] += red[t + s2];
    __syncthreads();
  }
  const float mu = red[0] / (float)DD;
  __syncthreads();
  const float d0 = ret[t] - mu, d1 = ret[t + 256] - mu;
  red[t] = d0 * d0 + d1 * d1;
  __syncthreads();
  for (int s2 = 128; s2 > 0; s2 >>= 1) {
    if (t < s2) red[t] += red[t + s2];
    __syncthreads();
  }
  const float rstd = rsqrtf(red[0] / (float)DD + 1e-5f);
  normed[b * DD + t]       = d0 * rstd * ln_g[t]       + ln_b[t];
  normed[b * DD + t + 256] = d1 * rstd * ln_g[t + 256] + ln_b[t + 256];
}

// ---------------- kernel 3: out = normed @ Wo^T + bo -----------------------
__global__ __launch_bounds__(128) void outproj_kernel(
    const float* __restrict__ normed, const float* __restrict__ Wo,
    const float* __restrict__ bo, float* __restrict__ out) {
  const int b     = blockIdx.x >> 2;
  const int dbase = (blockIdx.x & 3) * 128;
  const int t     = threadIdx.x;
  __shared__ float sN[DD];
  for (int i = t; i < DD; i += 128) sN[i] = normed[b * DD + i];
  __syncthreads();
  const int d = dbase + t;
  const float* wr = Wo + (long)d * DD;
  float acc = 0.0f;
#pragma unroll 4
  for (int j = 0; j < DD; ++j) acc += sN[j] * wr[j];
  out[b * DD + d] = acc + bo[d];
}

// ---------------------------------------------------------------------------
extern "C" void kernel_launch(void* const* d_in, const int* in_sizes, int n_in,
                              void* d_out, int out_size, void* d_ws, size_t ws_size,
                              hipStream_t stream) {
  const float* keys   = (const float*)d_in[0];
  const float* values = (const float*)d_in[1];
  const float* query  = (const float*)d_in[2];
  const float* Wbk    = (const float*)d_in[3];
  const float* bbk    = (const float*)d_in[4];
  const float* Wbv    = (const float*)d_in[5];
  const float* bbv    = (const float*)d_in[6];
  const float* Wbc    = (const float*)d_in[7];
  const float* bbc    = (const float*)d_in[8];
  const float* Wuq    = (const float*)d_in[9];
  const float* buq    = (const float*)d_in[10];
  const float* Wue    = (const float*)d_in[11];
  const float* bue    = (const float*)d_in[12];
  const float* ln_g   = (const float*)d_in[13];
  const float* ln_b   = (const float*)d_in[14];
  const float* Wo     = (const float*)d_in[15];
  const float* bo     = (const float*)d_in[16];
  float* out = (float*)d_out;

  // workspace layout (256B-aligned slices)
  char* w = (char*)d_ws;
  float*    S      = (float*)(w);                       //  2 KB : S[8][64]
  _Float16* wk16   = (_Float16*)(w + 2048);             // 64 KB : Wbk f16
  _Float16* wv16   = (_Float16*)(w + 2048 + 65536);     // 64 KB : Wbv f16
  float*    normed = (float*)(w + 2048 + 131072);       // 16 KB : normed[8][512]

  prep_kernel<<<128, 256, 0, stream>>>(Wbk, Wbv, wk16, wv16, S);
  bind_reduce_kernel<<<256, 256, 0, stream>>>(keys, values, wk16, wv16, bbk, bbv, S);
  epilogue_kernel<<<NB, 256, 0, stream>>>(S, query, Wbc, bbc, Wuq, buq,
                                          Wue, bue, ln_g, ln_b, normed);
  outproj_kernel<<<NB * 4, 128, 0, stream>>>(normed, Wo, bo, out);
}
